// MultiHeadAttention_3599182594845
// MI455X (gfx1250) — matrix operand order repaired and checked by reading, not compiled, not run on hardware
//
#include <hip/hip_runtime.h>
#include <hip/hip_bf16.h>

// ---------------------------------------------------------------------------
// MultiHeadAttention for MI455X (gfx1250): f16 WMMA pipeline
//   x[8,2048,384] f32 -> Q,K,V per head -> causal flash attention -> out proj
// Round 2: 32-query flash tiles (2x K/V fragment reuse), 32x64 proj tiles,
//          prefetch of next K/V step.
// ---------------------------------------------------------------------------

typedef __attribute__((ext_vector_type(16))) _Float16 v16h;
typedef __attribute__((ext_vector_type(8)))  _Float16 v8h;
typedef __attribute__((ext_vector_type(8)))  float    v8f;

constexpr int Bb = 8;
constexpr int Tt = 2048;
constexpr int Cd = 384;
constexpr int Hh = 6;
constexpr int Dd = 64;

union FragU { v16h v; v8h h[2]; };

__device__ __forceinline__ v8f wmma32(v16h a, v16h b, v8f c) {
  return __builtin_amdgcn_wmma_f32_16x16x32_f16(
      /*neg_a=*/false, a, /*neg_b=*/false, b,
      /*c_mod=*/(short)0, c, /*reuse_a=*/false, /*reuse_b=*/false);
}

// A-matrix fragment (16x32, f16) from row-major [rows][ld] at (row0,k0).
// Lane L holds row row0+(L&15); K values {0..7,16..23} (L<16) / {8..15,24..31}.
__device__ __forceinline__ v16h load_frag_a(const _Float16* __restrict__ base,
                                            int row0, int k0, int ld, int lane) {
  const int half = lane >> 4;
  const _Float16* p = base + (size_t)(row0 + (lane & 15)) * ld + k0 + 8 * half;
  FragU f;
  f.h[0] = *(const v8h*)(p);
  f.h[1] = *(const v8h*)(p + 16);
  return f.v;
}

// B-matrix fragment (32x16, f16) from a row-major [N][K] matrix (ld = K stride)
// at (n0,k0). Lane L holds N=n0+(L&15); K values {0..7,16..23} (L<16) / {8..15,24..31}.
__device__ __forceinline__ v16h load_frag_b(const _Float16* __restrict__ base,
                                            int n0, int k0, int ld, int lane) {
  const _Float16* p = base + (size_t)(n0 + (lane & 15)) * ld + k0 + 8 * (lane >> 4);
  FragU f;
  f.h[0] = *(const v8h*)(p);
  f.h[1] = *(const v8h*)(p + 16);
  return f.v;
}

// ---------------------------------------------------------------------------
// Prep kernels
// ---------------------------------------------------------------------------
__global__ void mha_cvt_f16(const float* __restrict__ in, _Float16* __restrict__ out, int n) {
  int i = blockIdx.x * blockDim.x + threadIdx.x;
  if (i < n) out[i] = (_Float16)in[i];
}

// in[bt][r][c] (f32) -> out[bt][c][r] (f16)
__global__ void mha_transpose_f16(const float* __restrict__ in, _Float16* __restrict__ out,
                                  int batch, int R, int Cn) {
  int i = blockIdx.x * blockDim.x + threadIdx.x;
  int total = batch * R * Cn;
  if (i >= total) return;
  int c  = i % Cn;
  int r  = (i / Cn) % R;
  int bt = i / (Cn * R);
  out[(size_t)bt * R * Cn + (size_t)c * R + r] = (_Float16)in[i];
}

// ---------------------------------------------------------------------------
// Kernel 1: per-head QKV projection.  One wave per (b*h, 16-token tile).
//   Q16[bh][t][d] = (x @ Wq[h]) * Dh^-0.5     (row major)
//   K16[bh][t][d] =  x @ Wk[h]                (row major)
//   VT16[bh][d][t] = (x @ Wv[h])^T            (d major, for PV B-fragments)
// ---------------------------------------------------------------------------
__global__ __launch_bounds__(128) void mha_qkv_kernel(
    const _Float16* __restrict__ x16,
    const _Float16* __restrict__ wqT, const _Float16* __restrict__ wkT,
    const _Float16* __restrict__ wvT,
    _Float16* __restrict__ q16, _Float16* __restrict__ k16,
    _Float16* __restrict__ vT16) {
  const int lane = threadIdx.x & 31;
  const int wave = (blockIdx.x * blockDim.x + threadIdx.x) >> 5;
  const int bh = wave >> 7;          // Tt/16 = 128 tiles
  const int tt = wave & 127;
  const int b = bh / Hh, h = bh % Hh;

  const _Float16* xa = x16 + (size_t)b * Tt * Cd;
  const _Float16* wT[3] = { wqT + (size_t)h * Dd * Cd,
                            wkT + (size_t)h * Dd * Cd,
                            wvT + (size_t)h * Dd * Cd };

  v8f acc[3][4];
#pragma unroll
  for (int w = 0; w < 3; ++w)
#pragma unroll
    for (int nt = 0; nt < 4; ++nt)
      acc[w][nt] = (v8f){0.f,0.f,0.f,0.f,0.f,0.f,0.f,0.f};

#pragma unroll 4
  for (int kk = 0; kk < Cd / 32; ++kk) {
    v16h a = load_frag_a(xa, tt * 16, kk * 32, Cd, lane);
#pragma unroll
    for (int w = 0; w < 3; ++w) {
#pragma unroll
      for (int nt = 0; nt < 4; ++nt) {
        v16h bf = load_frag_b(wT[w], nt * 16, kk * 32, Cd, lane);
        acc[w][nt] = wmma32(a, bf, acc[w][nt]);
      }
    }
  }

  const int half = lane >> 4, n_ = lane & 15;
#pragma unroll
  for (int nt = 0; nt < 4; ++nt) {
#pragma unroll
    for (int r = 0; r < 8; ++r) {
      const int t = tt * 16 + r + 8 * half;
      const int d = nt * 16 + n_;
      q16[((size_t)bh * Tt + t) * Dd + d] = (_Float16)(acc[0][nt][r] * 0.125f); // Dh^-0.5
      k16[((size_t)bh * Tt + t) * Dd + d] = (_Float16)(acc[1][nt][r]);
      vT16[((size_t)bh * Dd + d) * Tt + t] = (_Float16)(acc[2][nt][r]);
    }
  }
}

// ---------------------------------------------------------------------------
// Kernel 2: causal flash attention.  One wave per (b*h, 32 query rows).
// Two 16-row M-tiles share every K/V B-fragment (2x reuse). Online softmax
// over 32-key steps; P bounced through wave-private LDS (C-layout->A-layout).
// ---------------------------------------------------------------------------
__global__ __launch_bounds__(128) void mha_flash_kernel(
    const _Float16* __restrict__ q16, const _Float16* __restrict__ k16,
    const _Float16* __restrict__ vT16, _Float16* __restrict__ att16) {
  __shared__ __align__(16) _Float16 Pl[4][32][40];   // 40: pad, keeps 16B align

  const int lane = threadIdx.x & 31;
  const int wid  = threadIdx.x >> 5;
  const int wave = blockIdx.x * 4 + wid;
  const int bh = wave >> 6;          // Tt/32 = 64 query groups
  const int qg = wave & 63;
  const int b = bh / Hh, h = bh % Hh;
  const int qbase = qg * 32;
  const int half = lane >> 4, n_ = lane & 15;

  const _Float16* qb = q16 + (size_t)bh * Tt * Dd;
  const _Float16* kb = k16 + (size_t)bh * Tt * Dd;
  const _Float16* vb = vT16 + (size_t)bh * Dd * Tt;

  v16h aq[2][2];
#pragma unroll
  for (int mt = 0; mt < 2; ++mt) {
    aq[mt][0] = load_frag_a(qb, qbase + mt * 16, 0, Dd, lane);
    aq[mt][1] = load_frag_a(qb, qbase + mt * 16, 32, Dd, lane);
  }

  v8f o[2][4];
#pragma unroll
  for (int mt = 0; mt < 2; ++mt)
#pragma unroll
    for (int nt = 0; nt < 4; ++nt)
      o[mt][nt] = (v8f){0.f,0.f,0.f,0.f,0.f,0.f,0.f,0.f};
  float mrow[2][8], lrow[2][8];
#pragma unroll
  for (int mt = 0; mt < 2; ++mt)
#pragma unroll
    for (int r = 0; r < 8; ++r) { mrow[mt][r] = -__builtin_inff(); lrow[mt][r] = 0.f; }

  const int nsteps = qg + 1;         // keys [0, qbase+32) in 32-wide steps
  for (int js = 0; js < nsteps; ++js) {
    const int s0 = js * 32;

    // prefetch next step's K rows and V columns while this step computes
    if (js + 1 < nsteps) {
      __builtin_prefetch(kb + (size_t)(s0 + 32) * Dd + lane * 64, 0, 0);
      __builtin_prefetch(vb + (size_t)lane * Tt + s0 + 32, 0, 0);
      __builtin_prefetch(vb + (size_t)(lane + 32) * Tt + s0 + 32, 0, 0);
    }

    // S tiles = Q(32x64) @ K^T(64x32): K fragments shared by both M-tiles
    v8f s[2][2];
#pragma unroll
    for (int st = 0; st < 2; ++st) {
      v16h b0 = load_frag_b(kb, s0 + st * 16, 0, Dd, lane);
      v16h b1 = load_frag_b(kb, s0 + st * 16, 32, Dd, lane);
#pragma unroll
      for (int mt = 0; mt < 2; ++mt) {
        v8f z = (v8f){0.f,0.f,0.f,0.f,0.f,0.f,0.f,0.f};
        z = wmma32(aq[mt][0], b0, z);
        s[mt][st] = wmma32(aq[mt][1], b1, z);
      }
    }

#pragma unroll
    for (int mt = 0; mt < 2; ++mt) {
      // causal mask: element (M,N): qglob = qbase+mt*16+M, kglob = s0+st*16+N
#pragma unroll
      for (int r = 0; r < 8; ++r) {
        const int qglob = qbase + mt * 16 + r + 8 * half;
#pragma unroll
        for (int st = 0; st < 2; ++st) {
          const int kglob = s0 + st * 16 + n_;
          s[mt][st][r] = (kglob > qglob) ? -1.0e30f : s[mt][st][r];
        }
      }

      // online softmax: row stats reduced across the 16-lane half-wave
      float alpha[8];
#pragma unroll
      for (int r = 0; r < 8; ++r) {
        float mx = fmaxf(s[mt][0][r], s[mt][1][r]);
#pragma unroll
        for (int m = 1; m < 16; m <<= 1) mx = fmaxf(mx, __shfl_xor(mx, m, 32));
        const float mn = fmaxf(mrow[mt][r], mx);
        alpha[r] = __expf(mrow[mt][r] - mn);
        mrow[mt][r] = mn;

        const float p0 = __expf(s[mt][0][r] - mn);
        const float p1 = __expf(s[mt][1][r] - mn);
        s[mt][0][r] = p0; s[mt][1][r] = p1;
        float rs = p0 + p1;
#pragma unroll
        for (int m = 1; m < 16; m <<= 1) rs += __shfl_xor(rs, m, 32);
        lrow[mt][r] = alpha[r] * lrow[mt][r] + rs;
      }

      // rescale accumulator
#pragma unroll
      for (int nt = 0; nt < 4; ++nt)
#pragma unroll
        for (int r = 0; r < 8; ++r) o[mt][nt][r] *= alpha[r];

      // P (C-layout f32) -> LDS row-major f16 [32][32+pad]
#pragma unroll
      for (int r = 0; r < 8; ++r) {
        const int M = mt * 16 + r + 8 * half;
        Pl[wid][M][n_]      = (_Float16)s[mt][0][r];
        Pl[wid][M][16 + n_] = (_Float16)s[mt][1][r];
      }
    }
    asm volatile("s_wait_dscnt 0" ::: "memory");

    // A-fragments of P from LDS (one per M-tile)
    FragU ap[2];
#pragma unroll
    for (int mt = 0; mt < 2; ++mt) {
      const _Float16* pp = &Pl[wid][mt * 16 + n_][8 * half];
      ap[mt].h[0] = *(const v8h*)(pp);
      ap[mt].h[1] = *(const v8h*)(pp + 16);
    }

    // O += P(32x32) @ V(32x64); each V fragment shared by both M-tiles
#pragma unroll
    for (int nt = 0; nt < 4; ++nt) {
      v16h bv = load_frag_b(vb, nt * 16, s0, Tt, lane);
#pragma unroll
      for (int mt = 0; mt < 2; ++mt)
        o[mt][nt] = wmma32(ap[mt].v, bv, o[mt][nt]);
    }
  }

  // epilogue: O / l, store concat-head f16 tile
#pragma unroll
  for (int mt = 0; mt < 2; ++mt) {
#pragma unroll
    for (int r = 0; r < 8; ++r) {
      const float inv = 1.0f / lrow[mt][r];
      const int t = qbase + mt * 16 + r + 8 * half;
#pragma unroll
      for (int nt = 0; nt < 4; ++nt) {
        const int d = nt * 16 + n_;
        att16[((size_t)b * Tt + t) * Cd + h * Dd + d] = (_Float16)(o[mt][nt][r] * inv);
      }
    }
  }
}

// ---------------------------------------------------------------------------
// Kernel 3: output projection  out = att16 @ Wp + bp  (fp32 out)
// One wave per (32-token tile, 64-col group); Wp fragments reused across the
// two M-tiles.
// ---------------------------------------------------------------------------
__global__ __launch_bounds__(128) void mha_proj_kernel(
    const _Float16* __restrict__ att16, const _Float16* __restrict__ wpT,
    const float* __restrict__ bp, float* __restrict__ out) {
  const int lane = threadIdx.x & 31;
  const int wave = (blockIdx.x * blockDim.x + threadIdx.x) >> 5;
  const int tt = wave / (Cd / 64);   // 32-token tiles
  const int ng = wave % (Cd / 64);

  v8f acc[2][4];
#pragma unroll
  for (int mt = 0; mt < 2; ++mt)
#pragma unroll
    for (int nt = 0; nt < 4; ++nt)
      acc[mt][nt] = (v8f){0.f,0.f,0.f,0.f,0.f,0.f,0.f,0.f};

#pragma unroll 4
  for (int kk = 0; kk < Cd / 32; ++kk) {
    v16h a0 = load_frag_a(att16, tt * 32,      kk * 32, Cd, lane);
    v16h a1 = load_frag_a(att16, tt * 32 + 16, kk * 32, Cd, lane);
#pragma unroll
    for (int nt = 0; nt < 4; ++nt) {
      v16h bf = load_frag_b(wpT, ng * 64 + nt * 16, kk * 32, Cd, lane);
      acc[0][nt] = wmma32(a0, bf, acc[0][nt]);
      acc[1][nt] = wmma32(a1, bf, acc[1][nt]);
    }
  }

  const int half = lane >> 4, n_ = lane & 15;
#pragma unroll
  for (int mt = 0; mt < 2; ++mt) {
#pragma unroll
    for (int nt = 0; nt < 4; ++nt) {
#pragma unroll
      for (int r = 0; r < 8; ++r) {
        const int tok = tt * 32 + mt * 16 + r + 8 * half;
        const int co  = ng * 64 + nt * 16 + n_;
        out[(size_t)tok * Cd + co] = acc[mt][nt][r] + bp[co];
      }
    }
  }
}

// ---------------------------------------------------------------------------
// Host launcher
// ---------------------------------------------------------------------------
extern "C" void kernel_launch(void* const* d_in, const int* in_sizes, int n_in,
                              void* d_out, int out_size, void* d_ws, size_t ws_size,
                              hipStream_t stream) {
  (void)in_sizes; (void)n_in; (void)out_size; (void)ws_size;
  const float* x  = (const float*)d_in[0];
  const float* Wq = (const float*)d_in[1];
  const float* Wk = (const float*)d_in[2];
  const float* Wv = (const float*)d_in[3];
  const float* Wp = (const float*)d_in[4];
  const float* bp = (const float*)d_in[5];
  float* out = (float*)d_out;

  char* ws = (char*)d_ws;
  size_t off = 0;
  auto take = [&](size_t elems) -> _Float16* {
    _Float16* p = (_Float16*)(ws + off);
    off = (off + elems * sizeof(_Float16) + 255) & ~(size_t)255;
    return p;
  };
  _Float16* x16   = take((size_t)Bb * Tt * Cd);
  _Float16* wqT   = take((size_t)Hh * Dd * Cd);
  _Float16* wkT   = take((size_t)Hh * Dd * Cd);
  _Float16* wvT   = take((size_t)Hh * Dd * Cd);
  _Float16* wpT   = take((size_t)Cd * Cd);
  _Float16* q16   = take((size_t)Bb * Hh * Tt * Dd);
  _Float16* k16   = take((size_t)Bb * Hh * Tt * Dd);
  _Float16* vT16  = take((size_t)Bb * Hh * Tt * Dd);
  _Float16* att16 = take((size_t)Bb * Tt * Cd);

  const int nX = Bb * Tt * Cd;
  mha_cvt_f16<<<(nX + 255) / 256, 256, 0, stream>>>(x, x16, nX);

  const int nW = Hh * Cd * Dd;
  mha_transpose_f16<<<(nW + 255) / 256, 256, 0, stream>>>(Wq, wqT, Hh, Cd, Dd);
  mha_transpose_f16<<<(nW + 255) / 256, 256, 0, stream>>>(Wk, wkT, Hh, Cd, Dd);
  mha_transpose_f16<<<(nW + 255) / 256, 256, 0, stream>>>(Wv, wvT, Hh, Cd, Dd);
  const int nP = Cd * Cd;
  mha_transpose_f16<<<(nP + 255) / 256, 256, 0, stream>>>(Wp, wpT, 1, Cd, Cd);

  // QKV: Bb*Hh*(Tt/16) waves, 4 waves / 128-thread block
  const int qkvBlocks = Bb * Hh * (Tt / 16) / 4;          // 1536
  mha_qkv_kernel<<<qkvBlocks, 128, 0, stream>>>(x16, wqT, wkT, wvT, q16, k16, vT16);

  // Flash: Bb*Hh*(Tt/32) waves
  const int flashBlocks = Bb * Hh * (Tt / 32) / 4;        // 768
  mha_flash_kernel<<<flashBlocks, 128, 0, stream>>>(q16, k16, vT16, att16);

  // Proj: (Bb*Tt/32)*(Cd/64) waves
  const int projBlocks = (Bb * Tt / 32) * (Cd / 64) / 4;  // 768
  mha_proj_kernel<<<projBlocks, 128, 0, stream>>>(att16, wpT, bp, out);
}
